// AutoPromptGenerator_13726715478803
// MI455X (gfx1250) — matrix operand order repaired and checked by reading, not compiled, not run on hardware
//
#include <hip/hip_runtime.h>

#define NB     4
#define NHEAD  12
#define HDIM   64
#define DMODEL 768
#define HW     1024
#define NHT    (NB * NHEAD)   // 48 flattened heads

typedef _Float16 h16;
typedef __attribute__((ext_vector_type(16))) _Float16     v16h;
typedef __attribute__((ext_vector_type(8)))  float        v8f;
typedef __attribute__((ext_vector_type(4)))  unsigned int v4u;

union FragU { v16h v; v4u q[2]; };

// A-matrix fragment (16xK=32, f16). Per ISA 7.12.2:
// lanes 0-15: row m=lane, K = {0..7} U {16..23}; lanes 16-31: K = {8..15} U {24..31}.
// p points at [row0][k0] of a row-major (K-contiguous) tile; stride in elements.
__device__ __forceinline__ v16h frag_a(const h16* p, int stride) {
  const int lane = threadIdx.x & 31;
  const int lr = lane & 15, lh = lane >> 4;
  const h16* rp = p + lr * stride + (lh << 3);
  FragU f;
  f.q[0] = *(const v4u*)rp;
  f.q[1] = *(const v4u*)(rp + 16);
  return f.v;
}

// B-matrix fragment (K=32 x 16). Rows striped across lanes within a VGPR:
// lane n holds K = lane<16 ? {0..7} U {16..23} : {8..15} U {24..31} (2 per VGPR). Load from B^T stored
// row-major-in-K: two 16-byte chunks per lane at row n.
__device__ __forceinline__ v16h frag_b(const h16* p, int stride) {
  const int lane = threadIdx.x & 31;
  const int lr = lane & 15, lh = lane >> 4;
  const h16* rp = p + lr * stride + (lh << 3);
  FragU f;
  f.q[0] = *(const v4u*)rp;
  f.q[1] = *(const v4u*)(rp + 16);
  return f.v;
}

#define WMMA(a, b, c) \
  __builtin_amdgcn_wmma_f32_16x16x32_f16(false, (a), false, (b), (short)0, (c), false, false)

// ---------------- fp32 -> f16 convert ----------------
__global__ void cvt_f32_f16(const float* __restrict__ s, h16* __restrict__ d, int n) {
  int i = blockIdx.x * blockDim.x + threadIdx.x;
  if (i < n) d[i] = (h16)s[i];
}

// ---------------- WMMA GEMM: Y = X(Mx768) @ W(768x768) + b ----------------
// mode 0/1: scatter f16 to [head][tok][hd]   (Q, K)
// mode 2  : scatter f16 to [head][hd][tok]   (V transposed for P@V B-frags)
// mode 3  : fp32 to d_out [4096][768]        (final projection)
__global__ __launch_bounds__(256) void gemm_wmma(
    const h16* __restrict__ X, const h16* __restrict__ Wm,
    const float* __restrict__ bias, void* __restrict__ out, int mode)
{
  __shared__ __align__(16) h16 ldsA[64 * 40];   // 64 rows x 32 K (pad to 40)
  __shared__ __align__(16) h16 ldsB[64 * 40];   // B^T: 64 cols x 32 K (pad to 40)
  const int m0 = blockIdx.x << 6;
  const int n0 = blockIdx.y << 6;
  const int tid  = threadIdx.x;
  const int wid  = tid >> 5, lane = tid & 31;
  const int lr   = lane & 15, lh = lane >> 4;
  const int waveM = (wid >> 1) << 4;            // 0,16,32,48
  const int waveN = (wid & 1) << 5;             // 0,32

  v8f c0 = {}; v8f c1 = {};
  const int arow = tid >> 2, aseg = tid & 3;           // A: 64x32, uint4 each
  const int brow = tid >> 3, bseg = (tid & 7) << 3;    // B: read 8 n's of one k row

  for (int k0 = 0; k0 < DMODEL; k0 += 32) {
    v4u da = *(const v4u*)(X + (size_t)(m0 + arow) * DMODEL + k0 + (aseg << 3));
    v4u db = *(const v4u*)(Wm + (size_t)(k0 + brow) * DMODEL + n0 + bseg);
    *(v4u*)(&ldsA[arow * 40 + (aseg << 3)]) = da;
    union { v4u u; h16 e[8]; } tb; tb.u = db;
#pragma unroll
    for (int i = 0; i < 8; ++i) ldsB[(bseg + i) * 40 + brow] = tb.e[i];  // transpose
    __syncthreads();
    v16h a  = frag_a(&ldsA[waveM * 40], 40);
    v16h b0 = frag_b(&ldsB[waveN * 40], 40);
    v16h b1 = frag_b(&ldsB[(waveN + 16) * 40], 40);
    c0 = WMMA(a, b0, c0);
    c1 = WMMA(a, b1, c1);
    __syncthreads();
  }

  const int col0 = n0 + waveN + lr;
  const int col1 = col0 + 16;
  const float bv0 = bias[col0], bv1 = bias[col1];
  const int rbase = m0 + waveM + (lh << 3);     // C layout: m = v + 8*(lane/16)

  if (mode == 3) {
    float* O = (float*)out;
#pragma unroll
    for (int v = 0; v < 8; ++v) {
      O[(size_t)(rbase + v) * DMODEL + col0] = c0[v] + bv0;
      O[(size_t)(rbase + v) * DMODEL + col1] = c1[v] + bv1;
    }
  } else if (mode == 2) {
    h16* O = (h16*)out;
    const int head = col0 >> 6;                 // BN=64 == one head per block col
#pragma unroll
    for (int v = 0; v < 8; ++v) {
      int r = rbase + v, bb = r >> 10, t = r & 1023;
      size_t hb = (size_t)(bb * NHEAD + head) * HDIM * HW;
      O[hb + (size_t)(col0 & 63) * HW + t] = (h16)(c0[v] + bv0);
      O[hb + (size_t)(col1 & 63) * HW + t] = (h16)(c1[v] + bv1);
    }
  } else {
    h16* O = (h16*)out;
    const int head = col0 >> 6;
#pragma unroll
    for (int v = 0; v < 8; ++v) {
      int r = rbase + v, bb = r >> 10, t = r & 1023;
      size_t base = ((size_t)(bb * NHEAD + head) * HW + t) * HDIM;
      O[base + (col0 & 63)] = (h16)(c0[v] + bv0);
      O[base + (col1 & 63)] = (h16)(c1[v] + bv1);
    }
  }
}

// ---------------- decomposed rel-pos bias precompute ----------------
// horiz=0: out[head][q][kh] = sum_c qh[head][q][c] * rel_pos_h[(q/32)-kh+31][c]
// horiz=1: out[head][q][kw] = sum_c qh[head][q][c] * rel_pos_w[(q%32)-kw+31][c]
__global__ void rel_bias(const h16* __restrict__ qh, const float* __restrict__ table,
                         float* __restrict__ out, int horiz)
{
  int idx = blockIdx.x * blockDim.x + threadIdx.x;
  if (idx >= NHT * HW * 32) return;
  int k = idx & 31, q = (idx >> 5) & 1023, head = idx >> 15;
  int qc = horiz ? (q & 31) : (q >> 5);
  const h16*  qp = qh + ((size_t)head * HW + q) * HDIM;
  const float* tp = table + (size_t)(qc - k + 31) * HDIM;
  float s = 0.f;
#pragma unroll 8
  for (int c = 0; c < HDIM; ++c) s += (float)qp[c] * tp[c];
  out[idx] = s;
}

// ---------------- flash attention, 1 wave = 16 queries x all 1024 keys -----
__global__ __launch_bounds__(128) void attn_flash(
    const h16* __restrict__ Q, const h16* __restrict__ K, const h16* __restrict__ VT,
    const float* __restrict__ RH, const float* __restrict__ RW,
    h16* __restrict__ Out)
{
  __shared__ __align__(16) h16 ldsP[4][16 * 40];   // per-wave P-tile transpose buffer
  const int tid = threadIdx.x, wid = tid >> 5, lane = tid & 31;
  const int lr = lane & 15, lh = lane >> 4;
  const int head = blockIdx.x >> 4;
  const int q0 = (((blockIdx.x & 15) << 2) + wid) << 4;   // 16-query tile

  const h16* qbase = Q + ((size_t)head * HW + q0) * HDIM;
  const v16h aq0 = frag_a(qbase, HDIM);        // Q rows, c = 0..31
  const v16h aq1 = frag_a(qbase + 32, HDIM);   // c = 32..63

  // rel_w bias: key_w = t*16 + lr, loop-invariant -> preload 16 values/lane
  const int qrow0 = q0 + (lh << 3);            // C-layout row base for this lane
  float rw0[8], rw1[8];
#pragma unroll
  for (int v = 0; v < 8; ++v) {
    size_t rb = ((size_t)head * HW + qrow0 + v) * 32;
    rw0[v] = RW[rb + lr];
    rw1[v] = RW[rb + 16 + lr];
  }

  float m_i[8], l_i[8];
#pragma unroll
  for (int v = 0; v < 8; ++v) { m_i[v] = -1e30f; l_i[v] = 0.f; }
  v8f acc0 = {}, acc1 = {}, acc2 = {}, acc3 = {};
  const float scale = 0.125f;                  // 64^-0.5

  h16* P = ldsP[wid];
  const h16* vhead = VT + (size_t)head * HDIM * HW;

  for (int kc = 0; kc < 32; ++kc) {            // 32-key chunks
    const h16* kbase = K + ((size_t)head * HW + (kc << 5)) * HDIM;
    __builtin_prefetch(kbase + 32 * HDIM, 0, 1);
    v8f s0 = {}, s1 = {};
    s0 = WMMA(aq0, frag_b(kbase,             HDIM), s0);
    s0 = WMMA(aq1, frag_b(kbase + 32,        HDIM), s0);
    s1 = WMMA(aq0, frag_b(kbase + 16 * HDIM, HDIM), s1);
    s1 = WMMA(aq1, frag_b(kbase + 16 * HDIM + 32, HDIM), s1);

    float alpha[8];
#pragma unroll
    for (int v = 0; v < 8; ++v) {
      // key_h == kc for the whole chunk -> one rel_h load per row
      float rh = RH[((size_t)head * HW + qrow0 + v) * 32 + kc];
      float x0 = s0[v] * scale + rh + rw0[v];
      float x1 = s1[v] * scale + rh + rw1[v];
      s0[v] = x0; s1[v] = x1;
      float t = fmaxf(x0, x1);                 // row max across 16-lane N group
      t = fmaxf(t, __shfl_xor(t, 1));
      t = fmaxf(t, __shfl_xor(t, 2));
      t = fmaxf(t, __shfl_xor(t, 4));
      t = fmaxf(t, __shfl_xor(t, 8));
      float mn = fmaxf(m_i[v], t);
      float p0 = __expf(x0 - mn);
      float p1 = __expf(x1 - mn);
      s0[v] = p0; s1[v] = p1;
      float rs = p0 + p1;
      rs += __shfl_xor(rs, 1);
      rs += __shfl_xor(rs, 2);
      rs += __shfl_xor(rs, 4);
      rs += __shfl_xor(rs, 8);
      float a = __expf(m_i[v] - mn);
      l_i[v] = l_i[v] * a + rs;
      m_i[v] = mn;
      alpha[v] = a;
    }
#pragma unroll
    for (int v = 0; v < 8; ++v) {
      acc0[v] *= alpha[v]; acc1[v] *= alpha[v];
      acc2[v] *= alpha[v]; acc3[v] *= alpha[v];
      int m = v + (lh << 3);                   // C-layout -> A-layout via LDS
      P[m * 40 + lr]      = (h16)s0[v];
      P[m * 40 + 16 + lr] = (h16)s1[v];
    }
    asm volatile("s_wait_dscnt 0" ::: "memory");
    v16h ap = frag_a(P, 40);                   // P as 16x32 A matrix
    const h16* vbase = vhead + (kc << 5);      // VT[head][hd][key]
    acc0 = WMMA(ap, frag_b(vbase,           HW), acc0);
    acc1 = WMMA(ap, frag_b(vbase + 16 * HW, HW), acc1);
    acc2 = WMMA(ap, frag_b(vbase + 32 * HW, HW), acc2);
    acc3 = WMMA(ap, frag_b(vbase + 48 * HW, HW), acc3);
  }

  // epilogue: (B, HW, NH*HD) token-major f16 for the output projection
  const int bb = head / NHEAD, hl = head % NHEAD;
#pragma unroll
  for (int v = 0; v < 8; ++v) {
    int t = q0 + v + (lh << 3);
    size_t rowb = ((size_t)bb * HW + t) * DMODEL + hl * HDIM + lr;
    float inv = 1.f / l_i[v];
    Out[rowb]      = (h16)(acc0[v] * inv);
    Out[rowb + 16] = (h16)(acc1[v] * inv);
    Out[rowb + 32] = (h16)(acc2[v] * inv);
    Out[rowb + 48] = (h16)(acc3[v] * inv);
  }
}

extern "C" void kernel_launch(void* const* d_in, const int* in_sizes, int n_in,
                              void* d_out, int out_size, void* d_ws, size_t ws_size,
                              hipStream_t stream) {
  (void)in_sizes; (void)n_in; (void)out_size; (void)ws_size;
  const float* q   = (const float*)d_in[0];
  const float* k   = (const float*)d_in[1];
  const float* v   = (const float*)d_in[2];
  const float* Wq  = (const float*)d_in[3];
  const float* bq  = (const float*)d_in[4];
  const float* Wk  = (const float*)d_in[5];
  const float* bk  = (const float*)d_in[6];
  const float* Wv  = (const float*)d_in[7];
  const float* bv  = (const float*)d_in[8];
  const float* Wp  = (const float*)d_in[9];
  const float* bp  = (const float*)d_in[10];
  const float* rph = (const float*)d_in[11];
  const float* rpw = (const float*)d_in[12];

  char* w = (char*)d_ws;
  const size_t SZ_X = (size_t)NB * HW * DMODEL * sizeof(h16);  // 6 MB
  const size_t SZ_W = (size_t)DMODEL * DMODEL * sizeof(h16);   // 1.125 MB
  const size_t SZ_H = (size_t)NHT * HW * HDIM * sizeof(h16);   // 6 MB
  const size_t SZ_R = (size_t)NHT * HW * 32 * sizeof(float);   // 6 MB
  h16* q_h  = (h16*)w; w += SZ_X;
  h16* k_h  = (h16*)w; w += SZ_X;
  h16* v_h  = (h16*)w; w += SZ_X;
  h16* Wq_h = (h16*)w; w += SZ_W;
  h16* Wk_h = (h16*)w; w += SZ_W;
  h16* Wv_h = (h16*)w; w += SZ_W;
  h16* Wp_h = (h16*)w; w += SZ_W;
  h16* xq   = (h16*)w; w += SZ_H;   // [head][tok][hd]
  h16* xk   = (h16*)w; w += SZ_H;   // [head][key][hd]
  h16* xvT  = (h16*)w; w += SZ_H;   // [head][hd][key]
  h16* aout = (h16*)w; w += SZ_X;   // [B][tok][DMODEL]
  float* relh = (float*)w; w += SZ_R;
  float* relw = (float*)w; w += SZ_R;

  const int NX = NB * HW * DMODEL, NW = DMODEL * DMODEL;
  cvt_f32_f16<<<(NX + 255) / 256, 256, 0, stream>>>(q,  q_h,  NX);
  cvt_f32_f16<<<(NX + 255) / 256, 256, 0, stream>>>(k,  k_h,  NX);
  cvt_f32_f16<<<(NX + 255) / 256, 256, 0, stream>>>(v,  v_h,  NX);
  cvt_f32_f16<<<(NW + 255) / 256, 256, 0, stream>>>(Wq, Wq_h, NW);
  cvt_f32_f16<<<(NW + 255) / 256, 256, 0, stream>>>(Wk, Wk_h, NW);
  cvt_f32_f16<<<(NW + 255) / 256, 256, 0, stream>>>(Wv, Wv_h, NW);
  cvt_f32_f16<<<(NW + 255) / 256, 256, 0, stream>>>(Wp, Wp_h, NW);

  dim3 gg(NB * HW / 64, DMODEL / 64);   // (64, 12)
  gemm_wmma<<<gg, 256, 0, stream>>>(q_h, Wq_h, bq, xq,  0);
  gemm_wmma<<<gg, 256, 0, stream>>>(k_h, Wk_h, bk, xk,  1);
  gemm_wmma<<<gg, 256, 0, stream>>>(v_h, Wv_h, bv, xvT, 2);

  const int nrel = NHT * HW * 32;
  rel_bias<<<(nrel + 255) / 256, 256, 0, stream>>>(xq, rph, relh, 0);
  rel_bias<<<(nrel + 255) / 256, 256, 0, stream>>>(xq, rpw, relw, 1);

  attn_flash<<<NHT * 16, 128, 0, stream>>>(xq, xk, xvT, relh, relw, aout);

  gemm_wmma<<<gg, 256, 0, stream>>>(aout, Wp_h, bp, d_out, 3);
}
